// Batch_Attention_cos_22170621182644
// MI455X (gfx1250) — matrix operand order repaired and checked by reading, not compiled, not run on hardware
//
#include <hip/hip_runtime.h>
#include <hip/hip_bf16.h>

// Cosine-similarity attention, flash-style streaming (no materialized 8192x8192 score matrix).
// Precision path: f16 WMMA (v_wmma_f32_16x16x32_f16) with f32 accumulation.
// Softmax uses the fixed shift exp(s-1): cosine scores are bounded by 1, so no
// online max / output rescaling is needed -- accumulate numerator + denominator only.
// Round-1 refinements: half the Q A-fragments hoisted to registers (loop-invariant),
// double-buffered P tile -> one barrier per key iteration instead of two.

typedef __attribute__((ext_vector_type(16))) _Float16 v16h;
typedef __attribute__((ext_vector_type(8)))  _Float16 v8h;
typedef __attribute__((ext_vector_type(8)))  float    v8f;

#define B_ROWS 8192
#define C_DIM  512
#define BQ     32     // queries per workgroup
#define BK     64     // keys per iteration
#define SQ_LD  520    // padded LDS stride (halves) for Q tile (bank-conflict avoidance)
#define SP_LD  72     // padded LDS stride (halves) for P tile
#define LOG2E  1.44269504088896340736f

union AFrag { v16h v; v8h h[2]; };

// ---------------- prep: row norms, f16 Q/K operand (xn) and transposed f16 V ----------------
__global__ __launch_bounds__(256) void cosattn_prep(const float* __restrict__ x,
                                                    _Float16* __restrict__ xn,
                                                    _Float16* __restrict__ vt) {
  const int row = blockIdx.x;
  const int t   = threadIdx.x;           // 256 threads, 2 channels each
  const float* xr = x + (size_t)row * C_DIM;
  float v0 = xr[t];
  float v1 = xr[t + 256];
  float ss = v0 * v0 + v1 * v1;
  #pragma unroll
  for (int off = 16; off > 0; off >>= 1) ss += __shfl_xor(ss, off, 32);
  __shared__ float red[8];
  if ((t & 31) == 0) red[t >> 5] = ss;
  __syncthreads();
  const float tot = red[0] + red[1] + red[2] + red[3] + red[4] + red[5] + red[6] + red[7];
  const float inv = 1.0f / fmaxf(sqrtf(tot), 1e-12f);   // matches reference eps
  xn[(size_t)row * C_DIM + t]       = (_Float16)(v0 * inv);
  xn[(size_t)row * C_DIM + t + 256] = (_Float16)(v1 * inv);
  vt[(size_t)t * B_ROWS + row]         = (_Float16)v0;  // V^T: column-contiguous in key index
  vt[(size_t)(t + 256) * B_ROWS + row] = (_Float16)v1;
}

// ---------------- main: streaming attention, 8 waves / 32 queries per block ----------------
__global__ __launch_bounds__(256) void cosattn_main(const _Float16* __restrict__ xn,
                                                    const _Float16* __restrict__ vt,
                                                    float* __restrict__ out) {
  __shared__ _Float16 sQ[BQ * SQ_LD];       // 32 x 512 (padded) f16 query tile
  __shared__ _Float16 sP[2][BQ * SP_LD];    // double-buffered 32 x 64 (padded) f16 P tile
  __shared__ float    sDen[BQ];             // per-query softmax denominators

  const int tid  = threadIdx.x;
  const int wave = tid >> 5;
  const int lane = tid & 31;
  const int ln   = lane & 15;           // lane within half
  const int lh   = lane >> 4;           // which 16-lane half
  const int mt   = wave >> 2;           // S-phase: query sub-tile 0..1
  const int nt   = wave & 3;            // S-phase: key   sub-tile 0..3
  const int qbase = blockIdx.x * BQ;

  // Stage Q tile (already normalized, f16) into LDS: 1024 chunks of 8 halves.
  #pragma unroll
  for (int i = 0; i < 4; ++i) {
    const int chunk = tid + i * 256;
    const int off   = chunk * 8;
    const int r     = off >> 9;         // /512
    const int c     = off & 511;
    *(v8h*)&sQ[r * SQ_LD + c] = *(const v8h*)&xn[(size_t)(qbase + r) * C_DIM + c];
  }
  if (tid < BQ) sDen[tid] = 0.0f;
  __syncthreads();

  // Hoist the loop-invariant Q A-fragments for K = 0..255 into registers (8 x 8 VGPRs).
  // A layout: lane<16 -> K{+0..15}; lane>=16 -> K{+16..31}.
  const int am = mt * 16 + ln;
  AFrag qf[8];
  #pragma unroll
  for (int kk = 0; kk < 8; ++kk) {
    const int ko = kk * 32 + lh * 16;
    qf[kk].h[0] = *(const v8h*)&sQ[am * SQ_LD + ko];
    qf[kk].h[1] = *(const v8h*)&sQ[am * SQ_LD + ko + 8];
  }

  v8f o[2][4];                          // 32 rows x 64-col stripe of numerator, f32
  #pragma unroll
  for (int i = 0; i < 2; ++i)
    #pragma unroll
    for (int j = 0; j < 4; ++j)
      #pragma unroll
      for (int e = 0; e < 8; ++e) o[i][j][e] = 0.0f;
  float dsum[8];
  #pragma unroll
  for (int r = 0; r < 8; ++r) dsum[r] = 0.0f;

  int buf = 0;
  for (int k0 = 0; k0 < B_ROWS; k0 += BK, buf ^= 1) {
    _Float16* sPb = &sP[buf][0];
    // ---- S tile: 16x16 cosine scores, K=512 in 16 WMMA steps ----
    v8f c8;
    #pragma unroll
    for (int e = 0; e < 8; ++e) c8[e] = 0.0f;
    const int keyrow = k0 + nt * 16 + ln;
    if (k0 + BK < B_ROWS) {             // uniform branch: prefetch next key tile
      __builtin_prefetch(&xn[(size_t)(keyrow + BK) * C_DIM], 0, 0);
      __builtin_prefetch(&vt[(size_t)(wave * 64 + ln) * B_ROWS + k0 + BK], 0, 0);
    }
    const _Float16* kbase = &xn[(size_t)keyrow * C_DIM + lh * 16];
    #pragma unroll
    for (int kk = 0; kk < 8; ++kk) {    // K 0..255: A from registers
      v16h b = *(const v16h*)(kbase + kk * 32);
      c8 = __builtin_amdgcn_wmma_f32_16x16x32_f16(false, qf[kk].v, false, b, (short)0, c8, false, false);
    }
    #pragma unroll
    for (int kk = 8; kk < 16; ++kk) {   // K 256..511: A from LDS
      AFrag a;
      const int ko = kk * 32 + lh * 16;
      a.h[0] = *(const v8h*)&sQ[am * SQ_LD + ko];
      a.h[1] = *(const v8h*)&sQ[am * SQ_LD + ko + 8];
      v16h b = *(const v16h*)(kbase + kk * 32);
      c8 = __builtin_amdgcn_wmma_f32_16x16x32_f16(false, a.v, false, b, (short)0, c8, false, false);
    }
    // ---- p = exp(s - 1); cosine scores <= 1 so this is the stable softmax shift ----
    #pragma unroll
    for (int r = 0; r < 8; ++r) {
      const float p = exp2f((c8[r] - 1.0f) * LOG2E);
      dsum[r] += p;                     // per-wave partial denominator
      sPb[(mt * 16 + r + 8 * lh) * SP_LD + nt * 16 + ln] = (_Float16)p;
    }
    __syncthreads();                    // P write -> P read (single barrier; sP double-buffered)

    // ---- O += P . V : each wave owns a distinct 64-column output stripe ----
    const int cw = wave * 64;
    #pragma unroll
    for (int kb = 0; kb < BK; kb += 32) {
      AFrag pa0, pa1;
      const int pko = kb + lh * 16;
      pa0.h[0] = *(const v8h*)&sPb[(ln)      * SP_LD + pko];
      pa0.h[1] = *(const v8h*)&sPb[(ln)      * SP_LD + pko + 8];
      pa1.h[0] = *(const v8h*)&sPb[(16 + ln) * SP_LD + pko];
      pa1.h[1] = *(const v8h*)&sPb[(16 + ln) * SP_LD + pko + 8];
      const int krow = k0 + kb + lh * 16;
      #pragma unroll
      for (int ntile = 0; ntile < 4; ++ntile) {
        const int col = cw + ntile * 16 + ln;
        v16h bv = *(const v16h*)&vt[(size_t)col * B_ROWS + krow];  // V^T: contiguous in key
        o[0][ntile] = __builtin_amdgcn_wmma_f32_16x16x32_f16(false, pa0.v, false, bv, (short)0, o[0][ntile], false, false);
        o[1][ntile] = __builtin_amdgcn_wmma_f32_16x16x32_f16(false, pa1.v, false, bv, (short)0, o[1][ntile], false, false);
      }
    }
    // no trailing barrier: next iteration writes the other sP buffer; its pre-read
    // barrier guarantees every wave has finished this iteration's PV reads first.
  }

  // ---- reduce per-wave denominator partials: 16 lanes per half hold distinct columns ----
  #pragma unroll
  for (int r = 0; r < 8; ++r) {
    float s = dsum[r];
    s += __shfl_xor(s, 1, 32);
    s += __shfl_xor(s, 2, 32);
    s += __shfl_xor(s, 4, 32);
    s += __shfl_xor(s, 8, 32);
    if (ln == 0) atomicAdd(&sDen[mt * 16 + r + 8 * lh], s);   // lanes 0 & 16 -> rows r / r+8
  }
  __syncthreads();

  // ---- out = numerator / denominator ----
  const int cw = wave * 64;
  #pragma unroll
  for (int mtile = 0; mtile < 2; ++mtile) {
    #pragma unroll
    for (int r = 0; r < 8; ++r) {
      const int qrow = mtile * 16 + r + 8 * lh;
      const float rinv = 1.0f / sDen[qrow];
      #pragma unroll
      for (int ntile = 0; ntile < 4; ++ntile) {
        out[(size_t)(qbase + qrow) * C_DIM + cw + ntile * 16 + ln] = o[mtile][ntile][r] * rinv;
      }
    }
  }
}

extern "C" void kernel_launch(void* const* d_in, const int* in_sizes, int n_in,
                              void* d_out, int out_size, void* d_ws, size_t ws_size,
                              hipStream_t stream) {
  const float* x = (const float*)d_in[0];
  float* out = (float*)d_out;
  // Workspace layout: xn f16 [8192x512] (8 MB) | V^T f16 [512x8192] (8 MB)
  _Float16* xn = (_Float16*)d_ws;
  _Float16* vt = xn + (size_t)B_ROWS * C_DIM;
  cosattn_prep<<<B_ROWS, 256, 0, stream>>>(x, xn, vt);
  cosattn_main<<<B_ROWS / BQ, 256, 0, stream>>>(xn, vt, out);
}
